// scaled_dot_product_attention_72404558676789
// MI455X (gfx1250) — matrix operand order repaired and checked by reading, not compiled, not run on hardware
//
#include <hip/hip_runtime.h>
#include <hip/hip_bf16.h>

// ---------------------------------------------------------------------------
// Flash attention forward (causal), fp32 in/out, f16 WMMA compute, gfx1250.
// B=2 H=16 S=2048 D=64.
// Block = 128 threads = 4 wave32; each wave owns a 32-query tile (block covers
// 128 queries).  Keys processed 32/iteration via double-buffered LDS-staged
// f16 tiles: tile i+1 is staged while tile i is computed, with a single
// workgroup barrier per iteration, so HBM/L2 latency overlaps WMMA work.
// Per iteration per wave: 18x v_wmma_f32_16x16x32_f16 (8 QK^T + 8 PV + 2
// ones-column row-sum accumulators) sharing one set of K/V B-fragments.
// Causal mask applied only on the diagonal tile (kt == qw); kt < qw tiles take
// a maskless fast path; kt > qw skipped.  Softmax denominator is tracked as a
// WMMA accumulator against an all-ones B matrix (l = alpha*l + P.1), removing
// the cross-lane row-sum reduction.  Base-2 exp with log2(e) folded into Q.
// ---------------------------------------------------------------------------

typedef __attribute__((ext_vector_type(16))) _Float16 v16h;
typedef __attribute__((ext_vector_type(8)))  _Float16 v8h;
typedef __attribute__((ext_vector_type(2)))  _Float16 v2h;
typedef __attribute__((ext_vector_type(8)))  float    v8f;

#define S_LEN   2048
#define D_DIM   64
#define QSCALE  (0.125f * 1.4426950408889634f)   // (1/sqrt(64)) * log2(e)
#define NEG_BIG (-1.0e30f)
#define KT      32               // keys per iteration
#define WAVES   4                // 128 threads
#define QW      32               // queries per wave

static __device__ __forceinline__ v16h cat8(v8h lo, v8h hi) {
  return __builtin_shufflevector(lo, hi, 0,1,2,3,4,5,6,7,8,9,10,11,12,13,14,15);
}

static __device__ __forceinline__ v8f wmma16(v16h a, v16h b, v8f c) {
  // D = A(16x32) * B(32x16) + C, f32 accumulate
  return __builtin_amdgcn_wmma_f32_16x16x32_f16(false, a, false, b,
                                                (short)0, c, false, false);
}

static __device__ __forceinline__ float rowmax16(float v) {
  v = fmaxf(v, __shfl_xor(v, 1));
  v = fmaxf(v, __shfl_xor(v, 2));
  v = fmaxf(v, __shfl_xor(v, 4));
  v = fmaxf(v, __shfl_xor(v, 8));
  return v;
}

// Cooperative staging of one 32-key tile: K row-major f16, V transposed f16.
// Each thread handles two adjacent key rows x 8 dims so the transposed V pair
// (V[2k][d], V[2k+1][d]) packs into one v_cvt_pk_f16_f32 + ds_store_b32.
static __device__ __forceinline__ void stage_tiles(
    const float* __restrict__ Kb, const float* __restrict__ Vb,
    _Float16* __restrict__ Kl, _Float16* __restrict__ Vtl,
    const int kt, const int tid)
{
  const int kp = tid >> 3;          // key pair 0..15 -> keys 2kp, 2kp+1
  const int c  = (tid & 7) * 8;     // dim base 0..56
  const float* krow0 = Kb + (size_t)(kt + 2 * kp) * D_DIM + c;
  const float* krow1 = krow0 + D_DIM;
  v8h kv0, kv1;
  #pragma unroll
  for (int i = 0; i < 8; ++i) {
    kv0[i] = (_Float16)krow0[i];
    kv1[i] = (_Float16)krow1[i];
  }
  *reinterpret_cast<v8h*>(&Kl[(2 * kp)     * D_DIM + c]) = kv0;
  *reinterpret_cast<v8h*>(&Kl[(2 * kp + 1) * D_DIM + c]) = kv1;

  const float* vrow0 = Vb + (size_t)(kt + 2 * kp) * D_DIM + c;
  const float* vrow1 = vrow0 + D_DIM;
  #pragma unroll
  for (int i = 0; i < 8; ++i) {
    v2h pr;
    pr[0] = (_Float16)vrow0[i];
    pr[1] = (_Float16)vrow1[i];
    *reinterpret_cast<v2h*>(&Vtl[(c + i) * KT + 2 * kp]) = pr;
  }
}

// Online softmax over one 16-query x 32-key score pair (s0: keys kt..kt+15,
// s1: keys kt+16..kt+31).  Rescales the 4 output accumulators plus the
// row-sum accumulator and writes the f16 probability tile into wave-private
// LDS (rows rowbase..rowbase+15).  The row-sum itself is added later via a
// WMMA against an all-ones B matrix.
template <bool MASKED>
static __device__ __forceinline__ void online_softmax(
    v8f s0, v8f s1, float* __restrict__ m_i,
    v8f& o0, v8f& o1, v8f& o2, v8f& o3, v8f& osum,
    _Float16* __restrict__ Pw, const int rowbase,
    const int ln, const int hl, const int qsub, const int kt)
{
  const int key0 = kt + ln;
  const int key1 = kt + 16 + ln;
  #pragma unroll
  for (int r = 0; r < 8; ++r) {
    float a = s0[r], b = s1[r];
    if (MASKED) {
      const int q = qsub + r + 8 * hl;
      if (key0 > q) a = NEG_BIG;
      if (key1 > q) b = NEG_BIG;
    }
    const float mx    = rowmax16(fmaxf(a, b));
    const float mnew  = fmaxf(m_i[r], mx);
    const float alpha = __builtin_exp2f(m_i[r] - mnew);
    const float p0    = __builtin_exp2f(a - mnew);
    const float p1    = __builtin_exp2f(b - mnew);
    m_i[r] = mnew;
    o0[r] *= alpha; o1[r] *= alpha; o2[r] *= alpha; o3[r] *= alpha;
    osum[r] *= alpha;
    const int row = (rowbase + r + 8 * hl) * KT;
    Pw[row + ln]      = (_Float16)p0;
    Pw[row + 16 + ln] = (_Float16)p1;
  }
}

__global__ __launch_bounds__(WAVES * 32)
void fa_fwd_causal(const float* __restrict__ Qg, const float* __restrict__ Kg,
                   const float* __restrict__ Vg, float* __restrict__ Og)
{
  // double-buffered K/V tiles + per-wave P staging
  __shared__ __align__(16) _Float16 Klds [2][KT * D_DIM];     // [32 keys][64 dims]
  __shared__ __align__(16) _Float16 Vtlds[2][D_DIM * KT];     // [64 dims][32 keys]
  __shared__ __align__(16) _Float16 Plds [WAVES * QW * KT];

  const int tid  = threadIdx.x;
  const int w    = tid >> 5;
  const int lane = tid & 31;
  const int ln   = lane & 15;      // 0..15 within half-wave
  const int hl   = lane >> 4;      // 0 / 1

  const size_t bh = blockIdx.x;    // b*H + h
  const float* Qb = Qg + bh * (size_t)(S_LEN * D_DIM);
  const float* Kb = Kg + bh * (size_t)(S_LEN * D_DIM);
  const float* Vb = Vg + bh * (size_t)(S_LEN * D_DIM);
  float*       Ob = Og + bh * (size_t)(S_LEN * D_DIM);

  const int q0 = blockIdx.y * (WAVES * QW);   // first query of block
  const int qw = q0 + w * QW;                 // first query of this wave (mult of 32)

  // ---- Q -> A-fragments, two 16-query subtiles, exp2 scale folded in ----
  // A 16x32 f16 layout: lane holds row M=ln; halves cover K runs
  // [8*hl .. 8*hl+7] and [16+8*hl .. 16+8*hl+7].
  v16h aq0, aq1, aq2, aq3;
  {
    const float* qrowA = Qb + (size_t)(qw + ln) * D_DIM;
    const float* qrowB = Qb + (size_t)(qw + 16 + ln) * D_DIM;
    #pragma unroll
    for (int i = 0; i < 8; ++i) {
      aq0[i]     = (_Float16)(qrowA[      8 * hl + i] * QSCALE);
      aq0[i + 8] = (_Float16)(qrowA[16 +  8 * hl + i] * QSCALE);
      aq1[i]     = (_Float16)(qrowA[32 +  8 * hl + i] * QSCALE);
      aq1[i + 8] = (_Float16)(qrowA[48 +  8 * hl + i] * QSCALE);
      aq2[i]     = (_Float16)(qrowB[      8 * hl + i] * QSCALE);
      aq2[i + 8] = (_Float16)(qrowB[16 +  8 * hl + i] * QSCALE);
      aq3[i]     = (_Float16)(qrowB[32 +  8 * hl + i] * QSCALE);
      aq3[i + 8] = (_Float16)(qrowB[48 +  8 * hl + i] * QSCALE);
    }
  }

  // all-ones B fragment for the row-sum (softmax denominator) WMMAs
  v16h ones;
  #pragma unroll
  for (int i = 0; i < 16; ++i) ones[i] = (_Float16)1.0f;

  float mA[8], mB[8];
  v8f oA0 = {}, oA1 = {}, oA2 = {}, oA3 = {}, osumA = {};
  v8f oB0 = {}, oB1 = {}, oB2 = {}, oB3 = {}, osumB = {};
  #pragma unroll
  for (int r = 0; r < 8; ++r) { mA[r] = -3.0e38f; mB[r] = -3.0e38f; }

  _Float16* Pw = Plds + w * (QW * KT);

  const int kmax = q0 + WAVES * QW;   // causal bound for this block (mult of 32)

  // prologue: stage first tile into buffer 0
  stage_tiles(Kb, Vb, Klds[0], Vtlds[0], 0, tid);
  __syncthreads();

  int buf = 0;
  for (int kt = 0; kt < kmax; kt += KT, buf ^= 1) {
    // stage next tile into the other buffer while computing on this one;
    // global-load latency overlaps the WMMA/softmax work below.
    if (kt + KT < kmax)
      stage_tiles(Kb, Vb, Klds[buf ^ 1], Vtlds[buf ^ 1], kt + KT, tid);

    if (kt <= qw) {                  // kt > qw: fully masked tile, skip compute
      const _Float16* Kl  = Klds[buf];
      const _Float16* Vtl = Vtlds[buf];

      // ---- K B-fragments (shared by both query subtiles) ----
      // B 32x16 f16 layout: lane holds column N=ln; halves cover K runs
      // [8*hl .. 8*hl+7] and [16+8*hl .. 16+8*hl+7].
      v16h b0lo = cat8(*(const v8h*)&Kl[ ln       * D_DIM + 8 * hl],
                       *(const v8h*)&Kl[ ln       * D_DIM + 8 * hl + 16]);
      v16h b0hi = cat8(*(const v8h*)&Kl[ ln       * D_DIM + 32 + 8 * hl],
                       *(const v8h*)&Kl[ ln       * D_DIM + 32 + 8 * hl + 16]);
      v16h b1lo = cat8(*(const v8h*)&Kl[(16 + ln) * D_DIM + 8 * hl],
                       *(const v8h*)&Kl[(16 + ln) * D_DIM + 8 * hl + 16]);
      v16h b1hi = cat8(*(const v8h*)&Kl[(16 + ln) * D_DIM + 32 + 8 * hl],
                       *(const v8h*)&Kl[(16 + ln) * D_DIM + 32 + 8 * hl + 16]);

      v8f z = {};
      // ---- scores, subtile A (queries qw..qw+15) ----
      v8f sA0 = wmma16(aq0, b0lo, z);  sA0 = wmma16(aq1, b0hi, sA0);
      v8f sA1 = wmma16(aq0, b1lo, z);  sA1 = wmma16(aq1, b1hi, sA1);
      // ---- scores, subtile B (queries qw+16..qw+31) ----
      v8f sB0 = wmma16(aq2, b0lo, z);  sB0 = wmma16(aq3, b0hi, sB0);
      v8f sB1 = wmma16(aq2, b1lo, z);  sB1 = wmma16(aq3, b1hi, sB1);

      if (kt < qw) {
        // fully unmasked tile: maskless softmax fast path
        online_softmax<false>(sA0, sA1, mA, oA0, oA1, oA2, oA3, osumA,
                              Pw, 0,  ln, hl, qw,      kt);
        online_softmax<false>(sB0, sB1, mB, oB0, oB1, oB2, oB3, osumB,
                              Pw, 16, ln, hl, qw + 16, kt);
      } else {
        // diagonal tile (kt == qw): apply causal mask
        online_softmax<true>(sA0, sA1, mA, oA0, oA1, oA2, oA3, osumA,
                             Pw, 0,  ln, hl, qw,      kt);
        online_softmax<true>(sB0, sB1, mB, oB0, oB1, oB2, oB3, osumB,
                             Pw, 16, ln, hl, qw + 16, kt);
      }

      // ---- O += P(16x32) V(32x64), l += P.1: A-fragments from LDS ----
      v16h apA = cat8(*(const v8h*)&Pw[ ln       * KT +      8 * hl],
                      *(const v8h*)&Pw[ ln       * KT + 16 + 8 * hl]);
      v16h apB = cat8(*(const v8h*)&Pw[(16 + ln) * KT +      8 * hl],
                      *(const v8h*)&Pw[(16 + ln) * KT + 16 + 8 * hl]);

      v16h bv0 = cat8(*(const v8h*)&Vtl[( 0 + ln) * KT + 8 * hl],
                      *(const v8h*)&Vtl[( 0 + ln) * KT + 8 * hl + 16]);
      v16h bv1 = cat8(*(const v8h*)&Vtl[(16 + ln) * KT + 8 * hl],
                      *(const v8h*)&Vtl[(16 + ln) * KT + 8 * hl + 16]);
      v16h bv2 = cat8(*(const v8h*)&Vtl[(32 + ln) * KT + 8 * hl],
                      *(const v8h*)&Vtl[(32 + ln) * KT + 8 * hl + 16]);
      v16h bv3 = cat8(*(const v8h*)&Vtl[(48 + ln) * KT + 8 * hl],
                      *(const v8h*)&Vtl[(48 + ln) * KT + 8 * hl + 16]);

      osumA = wmma16(apA, ones, osumA);   // row sums: l = alpha*l + P.1
      osumB = wmma16(apB, ones, osumB);
      oA0 = wmma16(apA, bv0, oA0);
      oA1 = wmma16(apA, bv1, oA1);
      oA2 = wmma16(apA, bv2, oA2);
      oA3 = wmma16(apA, bv3, oA3);
      oB0 = wmma16(apB, bv0, oB0);
      oB1 = wmma16(apB, bv1, oB1);
      oB2 = wmma16(apB, bv2, oB2);
      oB3 = wmma16(apB, bv3, oB3);
    }

    // single barrier per iteration: publishes buffer buf^1 writes and retires
    // buffer buf reads before it is overwritten next iteration.
    __syncthreads();
  }

  // ---- normalize and store (16 consecutive lanes cover 16 dims of a row) ----
  #pragma unroll
  for (int r = 0; r < 8; ++r) {
    const float invA = 1.0f / osumA[r];   // every lane's column holds the row sum
    const float invB = 1.0f / osumB[r];
    float* orowA = Ob + (size_t)(qw + r + 8 * hl) * D_DIM;
    float* orowB = Ob + (size_t)(qw + 16 + r + 8 * hl) * D_DIM;
    orowA[ 0 + ln] = oA0[r] * invA;
    orowA[16 + ln] = oA1[r] * invA;
    orowA[32 + ln] = oA2[r] * invA;
    orowA[48 + ln] = oA3[r] * invA;
    orowB[ 0 + ln] = oB0[r] * invB;
    orowB[16 + ln] = oB1[r] * invB;
    orowB[32 + ln] = oB2[r] * invB;
    orowB[48 + ln] = oB3[r] * invB;
  }
}

extern "C" void kernel_launch(void* const* d_in, const int* in_sizes, int n_in,
                              void* d_out, int out_size, void* d_ws, size_t ws_size,
                              hipStream_t stream) {
  (void)n_in; (void)out_size; (void)d_ws; (void)ws_size;
  const float* Q = (const float*)d_in[0];
  const float* K = (const float*)d_in[1];
  const float* V = (const float*)d_in[2];
  // d_in[3] is the boolean causal mask; the kernel hardcodes causality.
  float* out = (float*)d_out;

  const int BH = in_sizes[0] / (S_LEN * D_DIM);        // B*H = 32
  dim3 grid((unsigned)BH, S_LEN / (WAVES * QW));       // (32, 16)
  fa_fwd_causal<<<grid, WAVES * 32, 0, stream>>>(Q, K, V, out);
}
